// BasicLSTM_65755949302179
// MI455X (gfx1250) — matrix operand order repaired and checked by reading, not compiled, not run on hardware
//
#include <hip/hip_runtime.h>
#include <hip/hip_bf16.h>
#include <math.h>

// ---------------------------------------------------------------------------
// BasicLSTM for MI455X (gfx1250, wave32, WMMA).
//
// S=128, B=64, I=H=O=1024, 32 decode steps.
// Cell step:  gates[64,4096] = [x|h][64,2048] @ [Wih|Whh]^T  + (b_ih+b_hh)
// Proj step:  logits[64,1024] = h @ W_out^T + b_out ; log_softmax rows
//
//  * One-time prep packs [Wih|Whh] and W_out into bf16 WMMA B-tile layout
//    (per-lane contiguous 32B chunks -> coalesced 1KB tile loads). 18MB of
//    bf16 weights stay resident in the 192MB L2 across all 159 steps.
//  * GEMM: one 16x16 M-tile x TWO N-tiles per wave (shared A fragment,
//    2 independent v_wmma_f32_16x16x32_bf16 per k-tile -> half the A traffic,
//    better WMMA pipelining), f32 accumulators, global_prefetch_b8 on B.
//  * Encode x-staging fused into the elementwise kernel (stage x[t+1] while
//    finishing step t) -> 128 fewer launches on the graph-captured chain.
// ---------------------------------------------------------------------------

typedef __attribute__((ext_vector_type(16))) __bf16 v16bf;
typedef __attribute__((ext_vector_type(8)))  __bf16 v8bf;
typedef __attribute__((ext_vector_type(8)))  float  v8f;

#define S_LEN 128
#define BATCH 64
#define HID   1024
#define GATES 4096           // 4*HID
#define KCAT  2048           // I + H
#define OUTL  32

// ---------------- weight packing --------------------------------------------
// B-matrix (K=32 x N=16, bf16) per-lane layout used here (contiguous k, same order as A):
//   lane l (0..31): column n = l%16, K-run = (l/16)*16 .. +16 (contiguous).
// Packed layout: tile index (nTile*kTiles + kTile), then lane, then 16 elems.

__global__ __launch_bounds__(256)
void pack_wcat_kernel(const float* __restrict__ Wih,
                      const float* __restrict__ Whh,
                      __bf16* __restrict__ dst) {
    // chunks = NT(256) * KT(64) * 32 lanes
    int chunk = blockIdx.x * blockDim.x + threadIdx.x;   // 0..524287
    int lane  = chunk & 31;
    int tile  = chunk >> 5;
    int kt    = tile & 63;            // KT = 64
    int nt    = tile >> 6;
    int n     = nt * 16 + (lane & 15);
    int kb    = kt * 32 + ((lane >> 4) << 4);
    __bf16* d = dst + (size_t)chunk * 16;
#pragma unroll
    for (int j = 0; j < 16; ++j) {
        int k = kb + j;
        float w = (k < HID) ? Wih[(size_t)n * HID + k]
                            : Whh[(size_t)n * HID + (k - HID)];
        d[j] = (__bf16)w;
    }
}

__global__ __launch_bounds__(256)
void pack_wout_kernel(const float* __restrict__ W, __bf16* __restrict__ dst) {
    // chunks = NT(64) * KT(32) * 32 lanes
    int chunk = blockIdx.x * blockDim.x + threadIdx.x;   // 0..65535
    int lane  = chunk & 31;
    int tile  = chunk >> 5;
    int kt    = tile & 31;            // KT = 32
    int nt    = tile >> 5;
    int n     = nt * 16 + (lane & 15);
    int kb    = kt * 32 + ((lane >> 4) << 4);
    __bf16* d = dst + (size_t)chunk * 16;
#pragma unroll
    for (int j = 0; j < 16; ++j)
        d[j] = (__bf16)W[(size_t)n * HID + kb + j];
}

__global__ __launch_bounds__(256)
void bias_sum_kernel(const float* __restrict__ bih,
                     const float* __restrict__ bhh,
                     float* __restrict__ bsum) {
    int n = blockIdx.x * blockDim.x + threadIdx.x;
    if (n < GATES) bsum[n] = bih[n] + bhh[n];
}

// ---------------- state init (also stages x0) -------------------------------

__global__ __launch_bounds__(256)
void init_state_kernel(const float* __restrict__ h0,
                       const float* __restrict__ c0,
                       const float* __restrict__ x0,
                       float* __restrict__ c,
                       float* __restrict__ h,
                       __bf16* __restrict__ xh) {
    int idx = blockIdx.x * blockDim.x + threadIdx.x;     // 0..65535
    int b = idx >> 10, j = idx & 1023;
    float hv = h0[idx];
    c[idx] = c0[idx];
    h[idx] = hv;
    xh[(size_t)b * KCAT + HID + j] = (__bf16)hv;
    xh[(size_t)b * KCAT + j]       = (__bf16)x0[idx];
}

// ---------------- WMMA GEMM: out[M=64, N] = A[M, K]bf16 @ Bpacked + bias ----
// One M-tile x two N-tiles per wave; 8 waves per block.
// A layout (bf16 16x32): lane m(=l&15): K 0..15 (l<16),
// K 16..31 (l>=16) -> two contiguous v8bf loads per k-tile.

__global__ __launch_bounds__(256)
void gemm_bf16_wmma(const __bf16* __restrict__ A, int lda, int kTiles,
                    const __bf16* __restrict__ Bp,
                    const float* __restrict__ bias,
                    float* __restrict__ out, int ldo) {
    const int lane   = threadIdx.x & 31;
    const int wave   = threadIdx.x >> 5;
    const int nTile0 = (blockIdx.x * 8 + wave) * 2;
    const int mTile  = blockIdx.y;

    const int mrow = mTile * 16 + (lane & 15);
    const __bf16* arow  = A + (size_t)mrow * lda + ((lane >> 4) << 4);
    const size_t tstr   = (size_t)kTiles * 512;          // elems per N-tile strip
    const __bf16* bptr0 = Bp + (size_t)nTile0 * tstr + lane * 16;
    const __bf16* bptr1 = bptr0 + tstr;

    v8f acc0 = {}, acc1 = {};
#pragma unroll 2
    for (int kt = 0; kt < kTiles; ++kt) {
        v8bf a0 = *(const v8bf*)(arow);
        v8bf a1 = *(const v8bf*)(arow + 8);
        v16bf a = __builtin_shufflevector(a0, a1,
                    0, 1, 2, 3, 4, 5, 6, 7, 8, 9, 10, 11, 12, 13, 14, 15);
        v16bf b0 = *(const v16bf*)(bptr0);
        v16bf b1 = *(const v16bf*)(bptr1);
        __builtin_prefetch(bptr0 + 512, 0, 1);   // global_prefetch_b8
        __builtin_prefetch(bptr1 + 512, 0, 1);
        acc0 = __builtin_amdgcn_wmma_f32_16x16x32_bf16(
                   false, a, false, b0, (short)0, acc0, false, false);
        acc1 = __builtin_amdgcn_wmma_f32_16x16x32_bf16(
                   false, a, false, b1, (short)0, acc1, false, false);
        arow  += 32;
        bptr0 += 512;
        bptr1 += 512;
    }

    // C/D layout: VGPR r -> row mTile*16 + r (+8 for lanes 16..31), col = l%16
    const int m0 = mTile * 16 + ((lane >> 4) << 3);
    const int n0 = nTile0 * 16 + (lane & 15);
    const int n1 = n0 + 16;
    const float bi0 = bias ? bias[n0] : 0.0f;
    const float bi1 = bias ? bias[n1] : 0.0f;
#pragma unroll
    for (int r = 0; r < 8; ++r) {
        out[(size_t)(m0 + r) * ldo + n0] = acc0[r] + bi0;
        out[(size_t)(m0 + r) * ldo + n1] = acc1[r] + bi1;
    }
}

// ---------------- elementwise LSTM gate update (+ stage next x) -------------

__global__ __launch_bounds__(256)
void lstm_elem_kernel(const float* __restrict__ gates,
                      float* __restrict__ c,
                      float* __restrict__ h,
                      __bf16* __restrict__ xh,
                      const float* __restrict__ next_x) {
    int idx = blockIdx.x * blockDim.x + threadIdx.x;     // 0..65535
    int b = idx >> 10, j = idx & 1023;
    const float* g = gates + (size_t)b * GATES;
    float gi = g[j];
    float gf = g[j + HID];
    float gg = g[j + 2 * HID];
    float go = g[j + 3 * HID];
    float si = 1.0f / (1.0f + expf(-gi));
    float sf = 1.0f / (1.0f + expf(-gf));
    float so = 1.0f / (1.0f + expf(-go));
    float cv = sf * c[idx] + si * tanhf(gg);
    float hv = so * tanhf(cv);
    c[idx] = cv;
    h[idx] = hv;
    xh[(size_t)b * KCAT + HID + j] = (__bf16)hv;
    if (next_x)                                   // encode: stage x[t+1]
        xh[(size_t)b * KCAT + j] = (__bf16)next_x[idx];
}

// ---------------- row log-softmax + feedback --------------------------------

__global__ __launch_bounds__(256)
void logsoftmax_kernel(const float* __restrict__ logits,
                       float* __restrict__ outSlice,
                       __bf16* __restrict__ xh) {
    __shared__ float red[256];
    const int b   = blockIdx.x;
    const int tid = threadIdx.x;
    const float* row = logits + (size_t)b * HID;

    float m = -INFINITY;
    for (int j = tid; j < HID; j += 256) m = fmaxf(m, row[j]);
    red[tid] = m; __syncthreads();
    for (int s = 128; s > 0; s >>= 1) {
        if (tid < s) red[tid] = fmaxf(red[tid], red[tid + s]);
        __syncthreads();
    }
    m = red[0]; __syncthreads();

    float sum = 0.0f;
    for (int j = tid; j < HID; j += 256) sum += expf(row[j] - m);
    red[tid] = sum; __syncthreads();
    for (int s = 128; s > 0; s >>= 1) {
        if (tid < s) red[tid] += red[tid + s];
        __syncthreads();
    }
    float ls = m + logf(red[0]);

    for (int j = tid; j < HID; j += 256) {
        float v = row[j] - ls;
        outSlice[(size_t)b * HID + j] = v;
        xh[(size_t)b * KCAT + j] = (__bf16)v;   // feedback as next x
    }
}

// ---------------- final h/c copy --------------------------------------------

__global__ __launch_bounds__(256)
void final_copy_kernel(const float* __restrict__ h,
                       const float* __restrict__ c,
                       float* __restrict__ out) {
    int idx = blockIdx.x * blockDim.x + threadIdx.x;     // 0..65535
    out[(size_t)OUTL * BATCH * HID + idx]       = h[idx];
    out[(size_t)(OUTL + 1) * BATCH * HID + idx] = c[idx];
}

// ---------------------------------------------------------------------------

extern "C" void kernel_launch(void* const* d_in, const int* in_sizes, int n_in,
                              void* d_out, int out_size, void* d_ws, size_t ws_size,
                              hipStream_t stream) {
    const float* input = (const float*)d_in[0];   // [S,B,I]
    const float* h0    = (const float*)d_in[1];   // [1,B,H]
    const float* c0    = (const float*)d_in[2];   // [1,B,H]
    const float* W_ih  = (const float*)d_in[3];   // [4H,I]
    const float* W_hh  = (const float*)d_in[4];   // [4H,H]
    const float* b_ih  = (const float*)d_in[5];   // [4H]
    const float* b_hh  = (const float*)d_in[6];   // [4H]
    const float* W_out = (const float*)d_in[7];   // [O,H]
    const float* b_out = (const float*)d_in[8];   // [O]
    float* out = (float*)d_out;

    // workspace layout (~21 MB)
    char* ws = (char*)d_ws;
    __bf16* WcatP  = (__bf16*)(ws);                       // 16,777,216
    __bf16* WoutP  = (__bf16*)(ws + 16777216);            //  2,097,152
    float*  bsum   = (float*) (ws + 18874368);            //     16,384
    float*  gates  = (float*) (ws + 18890752);            //  1,048,576
    float*  logits = (float*) (ws + 19939328);            //    262,144
    __bf16* xh     = (__bf16*)(ws + 20201472);            //    262,144
    float*  cbuf   = (float*) (ws + 20463616);            //    262,144
    float*  hbuf   = (float*) (ws + 20725760);            //    262,144

    // one-time prep: pack weights to bf16 WMMA tiles, fold biases, init state
    pack_wcat_kernel<<<2048, 256, 0, stream>>>(W_ih, W_hh, WcatP);
    pack_wout_kernel<<<256, 256, 0, stream>>>(W_out, WoutP);
    bias_sum_kernel<<<16, 256, 0, stream>>>(b_ih, b_hh, bsum);
    init_state_kernel<<<256, 256, 0, stream>>>(h0, c0, input, cbuf, hbuf, xh);

    const dim3 gCell(16, 4);   // NT=256 / (8 waves * 2 tiles), MT=4
    const dim3 gProj(4, 4);    // NT=64  / (8 waves * 2 tiles), MT=4

    // ---- encode: 128 LSTM steps (x[t] pre-staged by previous kernel) ------
    for (int t = 0; t < S_LEN; ++t) {
        gemm_bf16_wmma<<<gCell, 256, 0, stream>>>(xh, KCAT, 64, WcatP, bsum,
                                                  gates, GATES);
        const float* nx = (t + 1 < S_LEN) ? input + (size_t)(t + 1) * BATCH * HID
                                          : nullptr;
        lstm_elem_kernel<<<256, 256, 0, stream>>>(gates, cbuf, hbuf, xh, nx);
    }

    // ---- out0 = log_softmax(h @ W_out^T + b_out), also becomes x ----------
    gemm_bf16_wmma<<<gProj, 256, 0, stream>>>(xh + HID, KCAT, 32, WoutP, b_out,
                                              logits, HID);
    logsoftmax_kernel<<<BATCH, 256, 0, stream>>>(logits, out, xh);

    // ---- autoregressive decode: 31 steps ----------------------------------
    for (int t = 1; t < OUTL; ++t) {
        gemm_bf16_wmma<<<gCell, 256, 0, stream>>>(xh, KCAT, 64, WcatP, bsum,
                                                  gates, GATES);
        lstm_elem_kernel<<<256, 256, 0, stream>>>(gates, cbuf, hbuf, xh, nullptr);
        gemm_bf16_wmma<<<gProj, 256, 0, stream>>>(xh + HID, KCAT, 32, WoutP,
                                                  b_out, logits, HID);
        logsoftmax_kernel<<<BATCH, 256, 0, stream>>>(
            logits, out + (size_t)t * BATCH * HID, xh);
    }

    final_copy_kernel<<<256, 256, 0, stream>>>(hbuf, cbuf, out);
}
